// ScaledDotProductAttention_77077483094397
// MI455X (gfx1250) — matrix operand order repaired and checked by reading, not compiled, not run on hardware
//
#include <hip/hip_runtime.h>

typedef __attribute__((ext_vector_type(16))) _Float16 v16h;
typedef __attribute__((ext_vector_type(8)))  _Float16 v8h;
typedef __attribute__((ext_vector_type(8)))  float    v8f;

#define SHUF16(lo, hi) __builtin_shufflevector(lo, hi, 0,1,2,3,4,5,6,7,8,9,10,11,12,13,14,15)

namespace {
constexpr int Bsz = 2;
constexpr int Hn  = 16;
constexpr int Ssz = 2048;
constexpr int Dd  = 64;

constexpr int BLOCK_M = 128;  // query rows per workgroup (32 per wave, 2 m-tiles)
constexpr int BLOCK_N = 64;   // key cols per iteration
constexpr int WAVES   = 4;
constexpr int THREADS = WAVES * 32;
constexpr int MTILES  = 2;    // 16-row query tiles per wave

constexpr int KSTRIDE = Dd + 8;       // halfs: [key][d], padded
constexpr int VSTRIDE = BLOCK_N + 8;  // halfs: [d][key], padded
constexpr int PSTRIDE = BLOCK_N + 8;  // halfs: [row][key], padded

constexpr float LOG2E  = 1.4426950408889634f;
constexpr float NEGBIG = -1.0e30f;
}

// Fused DPP16 row_xmask reductions: one VALU op per step, exchange+combine in the
// same instruction (DPP applies to src0; all source lanes valid inside each 16-lane row).
__device__ __forceinline__ float rowred_max(float x) {
    asm("v_max_num_f32_dpp %0, %0, %0 row_xmask:1 row_mask:0xf bank_mask:0xf bound_ctrl:1" : "+v"(x));
    asm("v_max_num_f32_dpp %0, %0, %0 row_xmask:2 row_mask:0xf bank_mask:0xf bound_ctrl:1" : "+v"(x));
    asm("v_max_num_f32_dpp %0, %0, %0 row_xmask:4 row_mask:0xf bank_mask:0xf bound_ctrl:1" : "+v"(x));
    asm("v_max_num_f32_dpp %0, %0, %0 row_xmask:8 row_mask:0xf bank_mask:0xf bound_ctrl:1" : "+v"(x));
    return x;
}
__device__ __forceinline__ float rowred_sum(float x) {
    asm("v_add_f32_dpp %0, %0, %0 row_xmask:1 row_mask:0xf bank_mask:0xf bound_ctrl:1" : "+v"(x));
    asm("v_add_f32_dpp %0, %0, %0 row_xmask:2 row_mask:0xf bank_mask:0xf bound_ctrl:1" : "+v"(x));
    asm("v_add_f32_dpp %0, %0, %0 row_xmask:4 row_mask:0xf bank_mask:0xf bound_ctrl:1" : "+v"(x));
    asm("v_add_f32_dpp %0, %0, %0 row_xmask:8 row_mask:0xf bank_mask:0xf bound_ctrl:1" : "+v"(x));
    return x;
}

__global__ __launch_bounds__(THREADS)
void fa_causal_wmma_kernel(const float* __restrict__ Q,
                           const float* __restrict__ K,
                           const float* __restrict__ V,
                           float* __restrict__ O)
{
    __shared__ _Float16 sK [BLOCK_N * KSTRIDE];      // K block, row-major [key][d]
    __shared__ _Float16 sVt[Dd      * VSTRIDE];      // V block, transposed [d][key]
    __shared__ _Float16 sP [WAVES * 16 * PSTRIDE];   // per-wave P tile [row][key]

    const int tid  = threadIdx.x;
    const int lane = tid & 31;
    const int wave = tid >> 5;
    const int hh   = lane >> 4;   // which 16-lane half
    const int ln   = lane & 15;

    const int head = blockIdx.y;                       // b*H + h
    const size_t hbase = (size_t)head * Ssz * Dd;
    const int q0 = blockIdx.x * BLOCK_M;
    const int qw0 = q0 + wave * (MTILES * 16);         // this wave's first query row

    // ---- load Q m-tiles as 16x32 f16 A-fragments, pre-scaled by 1/sqrt(D) ----
    // A layout: lane(0-15)=row, elems 0-7 -> K=kb..kb+7, elems 8-15 -> K=kb+16..kb+23, kb=hh*8
    v16h qa[MTILES][2];
    #pragma unroll
    for (int mt = 0; mt < MTILES; ++mt) {
        const float* qrow = Q + hbase + (size_t)(qw0 + mt * 16 + ln) * Dd;
        const int kb = hh * 8;
        #pragma unroll
        for (int kk = 0; kk < 2; ++kk) {
            const float* p0 = qrow + kk * 32 + kb;
            float t[16];
            *(float4*)&t[0]  = *(const float4*)(p0);
            *(float4*)&t[4]  = *(const float4*)(p0 + 4);
            *(float4*)&t[8]  = *(const float4*)(p0 + 16);
            *(float4*)&t[12] = *(const float4*)(p0 + 20);
            v16h f;
            #pragma unroll
            for (int e = 0; e < 16; ++e) f[e] = (_Float16)(t[e] * 0.125f);
            qa[mt][kk] = f;
        }
    }

    // ---- softmax state + output accumulators ----
    v8f o[MTILES][4];
    #pragma unroll
    for (int mt = 0; mt < MTILES; ++mt)
        #pragma unroll
        for (int t = 0; t < 4; ++t)
            #pragma unroll
            for (int r = 0; r < 8; ++r) o[mt][t][r] = 0.0f;

    float mrow[MTILES][8], lrow[MTILES][8];
    #pragma unroll
    for (int mt = 0; mt < MTILES; ++mt)
        #pragma unroll
        for (int r = 0; r < 8; ++r) { mrow[mt][r] = NEGBIG; lrow[mt][r] = 0.0f; }

    _Float16* sPw = &sP[wave * 16 * PSTRIDE];
    const int nblocks = 2 * blockIdx.x + 2;            // causal: keys [0, q0+128)

    for (int j = 0; j < nblocks; ++j) {
        const int jb = j * BLOCK_N;
        const float* Kp = K + hbase + (size_t)jb * Dd;
        const float* Vp = V + hbase + (size_t)jb * Dd;

        __syncthreads();
        // ---- stage K (row-major) and V (transposed) into LDS as f16 ----
        #pragma unroll
        for (int it = 0; it < 8; ++it) {
            const int idx = tid + it * THREADS;        // 0..1023 float4 slots
            const int row = idx >> 4;                  // key row 0..63
            const int c4  = (idx & 15) * 4;            // d col 0..60
            float4 kv = *(const float4*)(Kp + row * Dd + c4);
            _Float16* dk = &sK[row * KSTRIDE + c4];
            dk[0] = (_Float16)kv.x; dk[1] = (_Float16)kv.y;
            dk[2] = (_Float16)kv.z; dk[3] = (_Float16)kv.w;
            float4 vv = *(const float4*)(Vp + row * Dd + c4);
            sVt[(c4 + 0) * VSTRIDE + row] = (_Float16)vv.x;
            sVt[(c4 + 1) * VSTRIDE + row] = (_Float16)vv.y;
            sVt[(c4 + 2) * VSTRIDE + row] = (_Float16)vv.z;
            sVt[(c4 + 3) * VSTRIDE + row] = (_Float16)vv.w;
        }
        __syncthreads();

        if (j + 1 < nblocks) {  // pull next K/V block toward the caches
            __builtin_prefetch(Kp + BLOCK_N * Dd + tid * 32, 0, 1);
            __builtin_prefetch(Vp + BLOCK_N * Dd + tid * 32, 0, 1);
        }

        #pragma unroll
        for (int mt = 0; mt < MTILES; ++mt) {
            const int qw = qw0 + mt * 16;

            // ---- scores: S(16x64) = Q(16x64) x K^T, four 16x16 tiles ----
            // B layout: lane col N=ln, elems 0-7 -> K(d-dim) = kk*32 + hh*8 + e, elems 8-15 -> kk*32 + 16 + hh*8 + (e-8)
            v8f s[4];
            #pragma unroll
            for (int t = 0; t < 4; ++t) {
                v8f acc = {};
                #pragma unroll
                for (int kk = 0; kk < 2; ++kk) {
                    const _Float16* bp = &sK[(t * 16 + ln) * KSTRIDE + kk * 32 + hh * 8];
                    v8h blo = *(const v8h*)bp;
                    v8h bhi = *(const v8h*)(bp + 16);
                    v16h b = SHUF16(blo, bhi);
                    acc = __builtin_amdgcn_wmma_f32_16x16x32_f16(
                              false, qa[mt][kk], false, b, (short)0, acc, false, false);
                }
                s[t] = acc;
            }

            // ---- online softmax (C layout rows: m = r + 8*hh, col = ln + t*16) ----
            #pragma unroll
            for (int r = 0; r < 8; ++r) {
                const int qg = qw + r + 8 * hh;
                float v[4];
                #pragma unroll
                for (int t = 0; t < 4; ++t)
                    v[t] = (jb + t * 16 + ln <= qg) ? s[t][r] : NEGBIG;
                float mx = fmaxf(fmaxf(v[0], v[1]), fmaxf(v[2], v[3]));
                mx = rowred_max(mx);
                const float mnew  = fmaxf(mrow[mt][r], mx);
                const float alpha = __builtin_amdgcn_exp2f((mrow[mt][r] - mnew) * LOG2E);
                mrow[mt][r] = mnew;
                float p[4];
                #pragma unroll
                for (int t = 0; t < 4; ++t)
                    p[t] = __builtin_amdgcn_exp2f((v[t] - mnew) * LOG2E);
                float ps = (p[0] + p[1]) + (p[2] + p[3]);
                ps = rowred_sum(ps);
                lrow[mt][r] = lrow[mt][r] * alpha + ps;
                #pragma unroll
                for (int t = 0; t < 4; ++t) o[mt][t][r] *= alpha;
                // C layout -> row-major P tile in per-wave LDS
                #pragma unroll
                for (int t = 0; t < 4; ++t)
                    sPw[(r + 8 * hh) * PSTRIDE + t * 16 + ln] = (_Float16)p[t];
            }

            // cross-lane LDS round-trip within the wave: drain DS ops
            asm volatile("s_wait_dscnt 0" ::: "memory");

            // ---- P(16x64) as two A-fragments from LDS ----
            v16h pa[2];
            #pragma unroll
            for (int kk = 0; kk < 2; ++kk) {
                const _Float16* pp = &sPw[ln * PSTRIDE + kk * 32 + hh * 8];
                v8h plo = *(const v8h*)pp;           // k = kb..kb+7
                v8h phi = *(const v8h*)(pp + 16);    // k = kb+16..kb+23
                pa[kk] = SHUF16(plo, phi);
            }

            // ---- O += P x V : four 16x16 d-tiles, two k-steps each ----
            #pragma unroll
            for (int t = 0; t < 4; ++t) {
                #pragma unroll
                for (int kk = 0; kk < 2; ++kk) {
                    const _Float16* vp = &sVt[(t * 16 + ln) * VSTRIDE + kk * 32 + hh * 8];
                    v8h vlo = *(const v8h*)vp;
                    v8h vhi = *(const v8h*)(vp + 16);
                    v16h vb = SHUF16(vlo, vhi);
                    o[mt][t] = __builtin_amdgcn_wmma_f32_16x16x32_f16(
                                   false, pa[kk], false, vb, (short)0, o[mt][t], false, false);
                }
            }
        }
    }

    // ---- epilogue: normalize and store fp32 ----
    #pragma unroll
    for (int mt = 0; mt < MTILES; ++mt) {
        #pragma unroll
        for (int r = 0; r < 8; ++r) {
            const float inv = __builtin_amdgcn_rcpf(lrow[mt][r]);
            const int row = qw0 + mt * 16 + r + 8 * hh;
            float* op = O + hbase + (size_t)row * Dd + ln;
            op[0]  = o[mt][0][r] * inv;
            op[16] = o[mt][1][r] * inv;
            op[32] = o[mt][2][r] * inv;
            op[48] = o[mt][3][r] * inv;
        }
    }
}

extern "C" void kernel_launch(void* const* d_in, const int* in_sizes, int n_in,
                              void* d_out, int out_size, void* d_ws, size_t ws_size,
                              hipStream_t stream) {
    (void)in_sizes; (void)n_in; (void)out_size; (void)d_ws; (void)ws_size;
    const float* Q = (const float*)d_in[0];
    const float* K = (const float*)d_in[1];
    const float* V = (const float*)d_in[2];
    // d_in[3] is the causal bool mask; the kernel applies causality analytically.
    float* O = (float*)d_out;
    dim3 grid(Ssz / BLOCK_M, Bsz * Hn);
    fa_causal_wmma_kernel<<<grid, dim3(THREADS), 0, stream>>>(Q, K, V, O);
}
